// ScanAttention_66288525246575
// MI455X (gfx1250) — matrix operand order repaired and checked by reading, not compiled, not run on hardware
//
#include <hip/hip_runtime.h>
#include <hip/hip_bf16.h>

#define DIM        512
#define NUM_HEADS  8
#define HEAD_DIM   64
#define BATCH      2
#define TLEN       1024
#define ROWS       (BATCH * TLEN)      // 2048
#define EPSV       1e-5f
#define CHUNKS     16
#define CHUNK_LEN  (TLEN / CHUNKS)     // 64
#define NEG_BIG    (-3.0e38f)

typedef __attribute__((ext_vector_type(16))) _Float16 v16h;
typedef __attribute__((ext_vector_type(8)))  float    v8f;
typedef __attribute__((ext_vector_type(4)))  unsigned int v4u;
typedef __attribute__((ext_vector_type(8)))  unsigned int v8u;

__device__ __forceinline__ float softplus_f(float x, float beta, float inv_beta) {
    float bx = beta * x;
    float sp = (bx > 20.f) ? bx : log1pf(__expf(bx));
    return inv_beta * sp;
}

__device__ __forceinline__ float logaddexp_f(float a, float b) {
    float mx = fmaxf(a, b);
    float mn = fminf(a, b);
    return mx + log1pf(__expf(mn - mx));
}

// ---------------------------------------------------------------------------
// Tensor Data Mover: 2D tile (rows x cols fp32, row stride in elements) from
// global memory into LDS.  Descriptor per CDNA5 ISA ch.8 (D# groups 0/1, 2D
// tile, data_size=4B, no padding/iteration/multicast).  Issue from ONE wave
// (EXEC is ignored by TDM; gating avoids duplicate DMAs), then
// s_wait_tensorcnt 0 before the workgroup barrier publishes the LDS tile.
// ---------------------------------------------------------------------------
__device__ __forceinline__ void tdm_load_2d(unsigned int lds_off, const float* gptr,
                                            unsigned int rows, unsigned int cols,
                                            unsigned int row_stride_elems)
{
    unsigned long long ga = (unsigned long long)(uintptr_t)gptr;
    v4u g0;
    g0[0] = 1u;                                                 // count=1, user mode
    g0[1] = lds_off;                                            // lds_addr (bytes)
    g0[2] = (unsigned int)(ga & 0xFFFFFFFFu);                   // global_addr lo
    g0[3] = (unsigned int)((ga >> 32) & 0x1FFFFFFu) | (2u << 30); // addr hi | type=2
    v8u g1;
    g1[0] = (2u << 16);                                         // data_size=2 (4B)
    g1[1] = (cols & 0xFFFFu) << 16;                             // tensor_dim0 lo16
    g1[2] = ((cols >> 16) & 0xFFFFu) | ((rows & 0xFFFFu) << 16); // dim0 hi | dim1 lo
    g1[3] = ((rows >> 16) & 0xFFFFu) | ((cols & 0xFFFFu) << 16); // dim1 hi | tile_dim0
    g1[4] = (rows & 0xFFFFu);                                   // tile_dim1 (tile_dim2=0)
    g1[5] = row_stride_elems;                                   // tensor_dim0_stride lo32
    g1[6] = 0u;                                                 // stride hi | dim1_stride lo
    g1[7] = 0u;
    asm volatile("tensor_load_to_lds %0, %1" :: "s"(g0), "s"(g1) : "memory");
}

// ---------------------------------------------------------------------------
// WMMA GEMM: O[wsel] = A(ROWSx512) @ W[wsel](512x512) + bias[wsel]
// One 16x16 output tile per wave; K-loop over 512 in steps of 32 using
// v_wmma_f32_16x16x32_f16 with fp32 accumulation.
// ---------------------------------------------------------------------------
__global__ __launch_bounds__(256)
void gemm512_wmma(const float* __restrict__ A,
                  const float* __restrict__ W0, const float* __restrict__ W1,
                  const float* __restrict__ W2,
                  const float* __restrict__ b0, const float* __restrict__ b1,
                  const float* __restrict__ b2,
                  float* __restrict__ O0, float* __restrict__ O1,
                  float* __restrict__ O2)
{
    const int lane = threadIdx.x & 31;
    const int wave = threadIdx.x >> 5;
    const int gN   = blockIdx.x * 128 + wave * 16;   // global column
    const int wsel = gN >> 9;
    const int locN = gN & 511;
    const float* W    = (wsel == 0) ? W0 : ((wsel == 1) ? W1 : W2);
    const float* bias = (wsel == 0) ? b0 : ((wsel == 1) ? b1 : b2);
    float*       O    = (wsel == 0) ? O0 : ((wsel == 1) ? O1 : O2);
    const int tileM = blockIdx.y * 16;

    // A-fragment (16-bit A 16x32, ISA 7.12.2): lanes 0-15 K 0-7/16-23,
    // lanes 16-31 K 8-15/24-31; M = lane&15.
    const int am   = lane & 15;
    const int asel = lane >> 4;
    // B-fragment (16-bit B 32x16): lanes 0-15 N=0..15 K 0-7/16-23, lanes 16-31 K 8-15/24-31.
    const int bn   = lane & 15;
    const int bko  = (lane >> 4) * 8;

    v8f acc = {};
    const float* arow = A + (size_t)(tileM + am) * DIM;

    #pragma unroll 4
    for (int k0 = 0; k0 < DIM; k0 += 32) {
        if (k0 + 32 < DIM) {
            __builtin_prefetch(arow + k0 + 32 + asel * 8, 0, 3);            // global_prefetch_b8
            __builtin_prefetch(W + (size_t)(k0 + 32 + bko) * DIM + locN + bn, 0, 3);
        }
        v16h a, b;
        const float* ap = arow + k0 + asel * 8;
        #pragma unroll
        for (int j = 0; j < 8; ++j) a[j]     = (_Float16)ap[j];
        #pragma unroll
        for (int j = 0; j < 8; ++j) a[8 + j] = (_Float16)ap[16 + j];
        const float* bp = W + (size_t)(k0 + bko) * DIM + locN + bn;
        #pragma unroll
        for (int j = 0; j < 8; ++j) b[j] = (_Float16)bp[(size_t)j * DIM];
        #pragma unroll
        for (int j = 0; j < 8; ++j) b[8 + j] = (_Float16)bp[(size_t)(16 + j) * DIM];

        acc = __builtin_amdgcn_wmma_f32_16x16x32_f16(
            false, a, false, b, (short)0, acc, false, false);
    }

    // C/D layout: VGPR r -> M = r (lanes 0-15) or r+8 (lanes 16-31), N = lane&15
    const int crow = tileM + (lane >> 4) * 8;
    const int ccol = locN + (lane & 15);
    const float bv = bias[ccol];
    #pragma unroll
    for (int r = 0; r < 8; ++r)
        O[(size_t)(crow + r) * DIM + ccol] = acc[r] + bv;
}

// ---------------------------------------------------------------------------
// Scan combine (reference _combine), state per (b,h):
//   m[64], D[64], N[64x64]   (N[p][n]; dx/dy broadcast along the p axis for N,
//   matching the reference's dx[..., None] with p == n)
// Thread mapping: n = tid & 63, p-group = tid >> 6; each thread holds
// N[pg*16+i][n] (i=0..15) in registers.
// ---------------------------------------------------------------------------

// Phase A: per-chunk aggregate (identity-seeded scan over CHUNK_LEN steps).
__global__ __launch_bounds__(256)
void scan_phaseA(const float* __restrict__ Kp, const float* __restrict__ Vp,
                 const float* __restrict__ log_beta, const float* __restrict__ log_temp,
                 float* __restrict__ aggN, float* __restrict__ aggM,
                 float* __restrict__ aggD)
{
    const int c   = blockIdx.x;
    const int bh  = blockIdx.y;
    const int b   = bh >> 3, h = bh & 7;
    const int tid = threadIdx.x;
    const int n   = tid & 63;
    const int pg  = tid >> 6;

    const float beta     = __expf(log_beta[h]);
    const float inv_beta = 1.f / beta;
    const float inv_temp = __expf(-log_temp[h]);

    __shared__ float Kl[CHUNK_LEN * 64];           // 16 KB chunk tile of K
    __shared__ float Vl[CHUNK_LEN * 64];           // 16 KB chunk tile of V
    __shared__ float m_s[64], D_s[64], Kf_s[64], dx_s[64], dy_s[64];

    const int t0 = c * CHUNK_LEN;
    const size_t tile0 = (size_t)(b * TLEN + t0) * DIM + h * HEAD_DIM;
    if (tid < 32) {                                // wave 0 drives the TDM
        tdm_load_2d((unsigned int)(uintptr_t)(void*)Kl, Kp + tile0,
                    CHUNK_LEN, HEAD_DIM, DIM);
        tdm_load_2d((unsigned int)(uintptr_t)(void*)Vl, Vp + tile0,
                    CHUNK_LEN, HEAD_DIM, DIM);
        __builtin_amdgcn_s_wait_tensorcnt(0);
    }
    float Nreg[16];
    #pragma unroll
    for (int i = 0; i < 16; ++i) Nreg[i] = 0.f;
    if (tid < 64) { m_s[n] = NEG_BIG; D_s[n] = 0.f; }
    __syncthreads();                               // publishes TDM tiles + init

    for (int tl = 0; tl < CHUNK_LEN; ++tl) {
        if (tid < 64) {
            float kv  = Kl[tl * 64 + n];
            float Kf  = softplus_f(kv * inv_temp, beta, inv_beta);
            float m0  = __logf(Kf + EPSV);
            float mo  = m_s[n];
            float mnw = fmaxf(mo, m0);
            float dx  = mo - mnw;
            float dy  = m0 - mnw;
            m_s[n]  = mnw;
            dx_s[n] = dx;  dy_s[n] = dy;
            Kf_s[n] = Kf;
            D_s[n]  = logaddexp_f(D_s[n] + dx, Kf + dy);
        }
        __syncthreads();
        const float Kfn = Kf_s[n];
        const float* vrow = &Vl[tl * 64 + pg * 16];
        #pragma unroll
        for (int i = 0; i < 16; ++i) {
            const int p = pg * 16 + i;
            Nreg[i] = logaddexp_f(Nreg[i] + dx_s[p], vrow[i] * Kfn + dy_s[p]);
        }
        __syncthreads();
    }

    const size_t base = (size_t)(bh * CHUNKS + c);
    #pragma unroll
    for (int i = 0; i < 16; ++i)
        aggN[base * 4096 + (size_t)(pg * 16 + i) * 64 + n] = Nreg[i];
    if (tid < 64) { aggM[base * 64 + n] = m_s[n]; aggD[base * 64 + n] = D_s[n]; }
}

// Phase B: exclusive scan of the CHUNKS aggregates per (b,h); in-place.
__global__ __launch_bounds__(256)
void scan_phaseB(float* __restrict__ aggN, float* __restrict__ aggM,
                 float* __restrict__ aggD)
{
    const int bh  = blockIdx.x;
    const int tid = threadIdx.x;
    const int n   = tid & 63;
    const int pg  = tid >> 6;

    __shared__ float Pm_s[64], dx_s[64], dy_s[64];
    float Preg[16];
    #pragma unroll
    for (int i = 0; i < 16; ++i) Preg[i] = 0.f;
    float Pd = 0.f;
    if (tid < 64) Pm_s[n] = NEG_BIG;
    __syncthreads();

    for (int c = 0; c < CHUNKS; ++c) {
        const size_t base = (size_t)(bh * CHUNKS + c);
        if (tid < 64) {
            float am = aggM[base * 64 + n];
            float ad = aggD[base * 64 + n];
            float pm = Pm_s[n];
            aggM[base * 64 + n] = pm;   // emit exclusive prefix
            aggD[base * 64 + n] = Pd;
            float mnw = fmaxf(pm, am);
            float dx  = pm - mnw;
            float dy  = am - mnw;
            dx_s[n] = dx; dy_s[n] = dy;
            Pm_s[n] = mnw;
            Pd = logaddexp_f(Pd + dx, ad + dy);
        }
        __syncthreads();
        #pragma unroll
        for (int i = 0; i < 16; ++i) {
            const int p = pg * 16 + i;
            const size_t idx = base * 4096 + (size_t)p * 64 + n;
            float av = aggN[idx];
            aggN[idx] = Preg[i];        // emit exclusive prefix
            Preg[i] = logaddexp_f(Preg[i] + dx_s[p], av + dy_s[p]);
        }
        __syncthreads();
    }
}

// Phase C: re-run each chunk seeded with its exclusive prefix; emit Y'.
__global__ __launch_bounds__(256)
void scan_phaseC(const float* __restrict__ Qp, const float* __restrict__ Kp,
                 const float* __restrict__ Vp,
                 const float* __restrict__ log_beta, const float* __restrict__ log_temp,
                 const float* __restrict__ log_sharp,
                 const float* __restrict__ aggN, const float* __restrict__ aggM,
                 const float* __restrict__ aggD,
                 float* __restrict__ Yp)
{
    const int c    = blockIdx.x;
    const int bh   = blockIdx.y;
    const int b    = bh >> 3, h = bh & 7;
    const int tid  = threadIdx.x;
    const int n    = tid & 63;
    const int pg   = tid >> 6;
    const int lane = tid & 31;

    const float beta     = __expf(log_beta[h]);
    const float inv_beta = 1.f / beta;
    const float inv_temp = __expf(-log_temp[h]);
    const float sharp    = __expf(log_sharp[h]);

    __shared__ float Ql[CHUNK_LEN * 64];
    __shared__ float Kl[CHUNK_LEN * 64];
    __shared__ float Vl[CHUNK_LEN * 64];
    __shared__ float m_s[64], D_s[64], Kf_s[64], Qf_s[64], dx_s[64], dy_s[64];
    __shared__ float red[256];
    __shared__ float ydenp[8];

    const int t0 = c * CHUNK_LEN;
    const size_t tile0 = (size_t)(b * TLEN + t0) * DIM + h * HEAD_DIM;
    if (tid < 32) {                                // wave 0 drives the TDM
        tdm_load_2d((unsigned int)(uintptr_t)(void*)Ql, Qp + tile0,
                    CHUNK_LEN, HEAD_DIM, DIM);
        tdm_load_2d((unsigned int)(uintptr_t)(void*)Kl, Kp + tile0,
                    CHUNK_LEN, HEAD_DIM, DIM);
        tdm_load_2d((unsigned int)(uintptr_t)(void*)Vl, Vp + tile0,
                    CHUNK_LEN, HEAD_DIM, DIM);
        __builtin_amdgcn_s_wait_tensorcnt(0);
    }

    const size_t base = (size_t)(bh * CHUNKS + c);
    float Nreg[16];
    #pragma unroll
    for (int i = 0; i < 16; ++i)
        Nreg[i] = aggN[base * 4096 + (size_t)(pg * 16 + i) * 64 + n];
    if (tid < 64) { m_s[n] = aggM[base * 64 + n]; D_s[n] = aggD[base * 64 + n]; }
    __syncthreads();                               // publishes TDM tiles + prefix

    for (int tl = 0; tl < CHUNK_LEN; ++tl) {
        if (tid < 64) {
            float kv  = Kl[tl * 64 + n];
            float qv  = Ql[tl * 64 + n];
            float Kf  = softplus_f(kv * inv_temp, beta, inv_beta);
            Qf_s[n]   = softplus_f(qv * inv_temp, beta, inv_beta);
            float m0  = __logf(Kf + EPSV);
            float mo  = m_s[n];
            float mnw = fmaxf(mo, m0);
            float dx  = mo - mnw;
            float dy  = m0 - mnw;
            m_s[n]  = mnw;
            dx_s[n] = dx;  dy_s[n] = dy;
            Kf_s[n] = Kf;
            D_s[n]  = logaddexp_f(D_s[n] + dx, Kf + dy);
        }
        __syncthreads();

        const float Kfn = Kf_s[n];
        const float* vrow = &Vl[tl * 64 + pg * 16];
        float part = 0.f;
        #pragma unroll
        for (int i = 0; i < 16; ++i) {
            const int p = pg * 16 + i;
            Nreg[i] = logaddexp_f(Nreg[i] + dx_s[p], vrow[i] * Kfn + dy_s[p]);
            part += Qf_s[p] * Nreg[i];   // Y_num[n] partial over this p-group
        }
        red[pg * 64 + n] = part;

        // Y_den = sum_n Qf[n]*D[n]: wave32 shuffle reduce (waves 0,1 cover n)
        float prod = (tid < 64) ? Qf_s[n] * D_s[n] : 0.f;
        #pragma unroll
        for (int off = 16; off > 0; off >>= 1)
            prod += __shfl_xor(prod, off, 32);
        if (lane == 0) ydenp[tid >> 5] = prod;
        __syncthreads();

        if (tid < 64) {
            float ynum = red[n] + red[64 + n] + red[128 + n] + red[192 + n];
            float yden = ydenp[0] + ydenp[1];
            float y = ynum / (yden + EPSV);
            y = copysignf(__powf(fabsf(y), sharp), y);
            y *= (float)(t0 + tl + 1) * (1.0f / (float)TLEN);
            Yp[tile0 + (size_t)tl * DIM + n] = y;
        }
        __syncthreads();
    }
}

// ---------------------------------------------------------------------------
extern "C" void kernel_launch(void* const* d_in, const int* in_sizes, int n_in,
                              void* d_out, int out_size, void* d_ws, size_t ws_size,
                              hipStream_t stream)
{
    const float* x  = (const float*)d_in[0];
    const float* wq = (const float*)d_in[1];
    const float* bq = (const float*)d_in[2];
    const float* wk = (const float*)d_in[3];
    const float* bk = (const float*)d_in[4];
    const float* wv = (const float*)d_in[5];
    const float* bv = (const float*)d_in[6];
    const float* wo = (const float*)d_in[7];
    const float* bo = (const float*)d_in[8];
    const float* lb = (const float*)d_in[9];
    const float* lt = (const float*)d_in[10];
    const float* ls = (const float*)d_in[11];
    float* out = (float*)d_out;
    float* ws  = (float*)d_ws;

    float* Q    = ws;
    float* K    = Q  + (size_t)ROWS * DIM;
    float* V    = K  + (size_t)ROWS * DIM;
    float* Yp   = V  + (size_t)ROWS * DIM;
    float* aggN = Yp + (size_t)ROWS * DIM;                            // 16*16*4096
    float* aggM = aggN + (size_t)BATCH * NUM_HEADS * CHUNKS * 4096;   // 16*16*64
    float* aggD = aggM + (size_t)BATCH * NUM_HEADS * CHUNKS * 64;

    // Fused QKV projection (WMMA)
    gemm512_wmma<<<dim3(3 * DIM / 128, ROWS / 16), 256, 0, stream>>>(
        x, wq, wk, wv, bq, bk, bv, Q, K, V);

    // Chunk-parallel associative scan (TDM-staged tiles)
    scan_phaseA<<<dim3(CHUNKS, BATCH * NUM_HEADS), 256, 0, stream>>>(
        K, V, lb, lt, aggN, aggM, aggD);
    scan_phaseB<<<BATCH * NUM_HEADS, 256, 0, stream>>>(aggN, aggM, aggD);
    scan_phaseC<<<dim3(CHUNKS, BATCH * NUM_HEADS), 256, 0, stream>>>(
        Q, K, V, lb, lt, ls, aggN, aggM, aggD, Yp);

    // Output projection (WMMA)
    gemm512_wmma<<<dim3(DIM / 128, ROWS / 16), 256, 0, stream>>>(
        Yp, wo, wo, wo, bo, bo, bo, out, out, out);
}
